// back_numGAT_40862318854481
// MI455X (gfx1250) — matrix operand order repaired and checked by reading, not compiled, not run on hardware
//
#include <hip/hip_runtime.h>

// ---------------------------------------------------------------------------
// GAT-style head, algebraically folded:
//   Wc = W_gc @ W1 (256x32), c0 = b_gc @ W1
//   Y       = inputs @ Wc                       (B,S,32)
//   seq_fts = adj @ Y + c0                      (B,S,32)   adj = graphs[:,3]
//   f1/f2   = seq_fts @ a1/a2                   (B,S)
//   m,s     = column-softmax stats over i of leaky(f1[i]+f2[j]) + bias[i,j]
//   ret     = leaky(P @ seq_fts), P regenerated on the fly; out = tile(ret,8)
// GEMMs: v_wmma_f32_16x16x32_bf16 (bf16 A/B, f32 accum).
// A-operand tiles of the two S-wide GEMMs are streamed through LDS with
// GLOBAL_LOAD_ASYNC_TO_LDS_B128 double buffering (ASYNCcnt / s_wait_asynccnt).
// ---------------------------------------------------------------------------

#define DEV __device__ __forceinline__

typedef __attribute__((ext_vector_type(16))) __bf16 v16bf;
typedef __attribute__((ext_vector_type(8)))  float  v8f;

constexpr int BB = 4, S = 2048, H = 256, HD = 32, HEADS = 8;

DEV float leaky(float x) { return x > 0.f ? x : 0.01f * x; }

DEV v8f wmma_bf16(v16bf a, v16bf b, v8f c) {
  return __builtin_amdgcn_wmma_f32_16x16x32_bf16(false, a, false, b,
                                                 (short)0, c, false, false);
}

// ---------------- async global->LDS tile streaming ----------------------
#if defined(__gfx1250__) &&                                                   \
    __has_builtin(__builtin_amdgcn_global_load_async_to_lds_b128) &&          \
    __has_builtin(__builtin_amdgcn_s_wait_asynccnt)
#define HAVE_ASYNC_LDS 1
#else
#define HAVE_ASYNC_LDS 0
#endif

// Builtin prototype (from clang diagnostic): first arg is
//   int __attribute__((vector_size(16))) addrspace(1)*   (global int4*)
// second is the LDS address (addrspace(3)), then imm offset, imm cpol.
typedef __attribute__((__vector_size__(4 * sizeof(int)))) int v4i_t;
typedef __attribute__((address_space(1))) v4i_t g_v4i;
typedef __attribute__((address_space(3))) v4i_t l_v4i;

// Each lane copies 64B (one half-row of a 16x32 f32 tile) global -> LDS.
DEV void tile_copy64(const float* gsrc, float* ldst) {
#if HAVE_ASYNC_LDS
  g_v4i* g = (g_v4i*)gsrc;
  l_v4i* l = (l_v4i*)ldst;
  __builtin_amdgcn_global_load_async_to_lds_b128(g, l, 0, 0);
  __builtin_amdgcn_global_load_async_to_lds_b128(g, l, 16, 0);
  __builtin_amdgcn_global_load_async_to_lds_b128(g, l, 32, 0);
  __builtin_amdgcn_global_load_async_to_lds_b128(g, l, 48, 0);
#else
  const float4* gs = (const float4*)gsrc;
  float4* ld = (float4*)ldst;
  ld[0] = gs[0];
  ld[1] = gs[1];
  ld[2] = gs[2];
  ld[3] = gs[3];
#endif
}

#if HAVE_ASYNC_LDS
#define WAIT_ASYNC(n)                                                         \
  do {                                                                        \
    __builtin_amdgcn_s_wait_asynccnt(n);                                      \
    __builtin_amdgcn_wave_barrier();                                          \
    asm volatile("" ::: "memory");                                            \
  } while (0)
#else
#define WAIT_ASYNC(n)                                                         \
  do {                                                                        \
    __builtin_amdgcn_wave_barrier();                                          \
    asm volatile("" ::: "memory");                                            \
  } while (0)
#endif

// ---------------- kernel 1: fold weights --------------------------------
// Wct[c][k] = (W_gc @ W1)[k][c]  (transposed, K-contiguous for B fragments)
__global__ void k_prep(const float* __restrict__ Wgc,
                       const float* __restrict__ bgc,
                       const float* __restrict__ W1,
                       float* __restrict__ Wct, float* __restrict__ c0) {
  int r = threadIdx.x;                       // 0..255 -> row of Wc
  float acc[HD];
#pragma unroll
  for (int c = 0; c < HD; ++c) acc[c] = 0.f;
  for (int k = 0; k < H; ++k) {
    float w = Wgc[r * H + k];
    const float* w1k = W1 + k * HD;
#pragma unroll
    for (int c = 0; c < HD; ++c) acc[c] += w * w1k[c];
  }
#pragma unroll
  for (int c = 0; c < HD; ++c) Wct[c * H + r] = acc[c];
  if (r < HD) {
    float a = 0.f;
    for (int k = 0; k < H; ++k) a += bgc[k] * W1[k * HD + r];
    c0[r] = a;
  }
}

// ---------------- kernel 2: Yt = (inputs @ Wc)^T in bf16 ----------------
__global__ void __launch_bounds__(256)
k_y(const float* __restrict__ inputs, const float* __restrict__ Wct,
    __bf16* __restrict__ Yt) {
  int lane = threadIdx.x & 31, wave = threadIdx.x >> 5;
  int tile = blockIdx.x * 8 + wave;          // 512 tiles = B * S/16
  int b = tile >> 7;
  int i0 = (tile & 127) << 4;
  int row = lane & 15, half = lane >> 4, col = lane & 15;
  const float* arow = inputs + ((size_t)b * S + i0 + row) * H;
  v8f acc0 = {}, acc1 = {};
  for (int k0 = 0; k0 < H; k0 += 32) {
    v16bf a, b0, b1;
#pragma unroll
    for (int v = 0; v < 8; ++v) {
      int kb = ((v >> 2) << 4) + (half << 3) + ((v & 3) << 1);   // A frag K
      a[2 * v]     = (__bf16)arow[k0 + kb];
      a[2 * v + 1] = (__bf16)arow[k0 + kb + 1];
      int kk = ((v >> 2) << 4) + (half << 3) + ((v & 3) << 1);   // B frag K
      const float* p0 = Wct + col * H + k0 + kk;
      const float* p1 = Wct + (col + 16) * H + k0 + kk;
      b0[2 * v] = (__bf16)p0[0]; b0[2 * v + 1] = (__bf16)p0[1];
      b1[2 * v] = (__bf16)p1[0]; b1[2 * v + 1] = (__bf16)p1[1];
    }
    acc0 = wmma_bf16(a, b0, acc0);
    acc1 = wmma_bf16(a, b1, acc1);
  }
#pragma unroll
  for (int r = 0; r < 8; ++r) {
    int gi = i0 + (half << 3) + r;
    Yt[((size_t)b * HD + col) * S + gi]      = (__bf16)acc0[r];
    Yt[((size_t)b * HD + col + 16) * S + gi] = (__bf16)acc1[r];
  }
}

// ---------------- kernel 3: seq_fts = adj @ Y + c0 ----------------------
// adj tiles (16x32 f32) double-buffered in LDS via async-to-LDS.
__global__ void __launch_bounds__(256)
k_seqfts(const float* __restrict__ graphs, const __bf16* __restrict__ Yt,
         const float* __restrict__ c0, float* __restrict__ sf,
         __bf16* __restrict__ sft) {
  __shared__ float atile[8][2][16 * 32];     // 32 KB
  int lane = threadIdx.x & 31, wave = threadIdx.x >> 5;
  int tile = blockIdx.x * 8 + wave;
  int b = tile >> 7;
  int i0 = (tile & 127) << 4;
  int row = lane & 15, half = lane >> 4, col = lane & 15;
  const float* adjb = graphs + ((size_t)b * 4 + 3) * (size_t)S * S;
  int crow = lane >> 1, cpart = lane & 1;    // copy mapping: 64B per lane
  const float* gsrc = adjb + (size_t)(i0 + crow) * S + cpart * 16;
  float* ldst0 = &atile[wave][0][crow * 32 + cpart * 16];
  float* ldst1 = &atile[wave][1][crow * 32 + cpart * 16];
  const __bf16* yb = Yt + (size_t)b * HD * S;
  v8f acc0 = {}, acc1 = {};

  tile_copy64(gsrc, ldst0);                  // prologue: tile 0 -> buf0
  for (int k0 = 0; k0 < S; k0 += 32) {
    if (k0 + 32 < S) {
      tile_copy64(gsrc + k0 + 32, (k0 & 32) ? ldst0 : ldst1);
      WAIT_ASYNC(4);                         // current tile resident
    } else {
      WAIT_ASYNC(0);
    }
    const float* cur = &atile[wave][(k0 >> 5) & 1][0];
    v16bf a, b0, b1;
#pragma unroll
    for (int v = 0; v < 8; ++v) {
      int kb = ((v >> 2) << 4) + (half << 3) + ((v & 3) << 1);
      a[2 * v]     = (__bf16)cur[row * 32 + kb];
      a[2 * v + 1] = (__bf16)cur[row * 32 + kb + 1];
      int kk = k0 + ((v >> 2) << 4) + (half << 3) + ((v & 3) << 1);
      const __bf16* p0 = yb + (size_t)col * S + kk;
      const __bf16* p1 = yb + (size_t)(col + 16) * S + kk;
      b0[2 * v] = p0[0]; b0[2 * v + 1] = p0[1];
      b1[2 * v] = p1[0]; b1[2 * v + 1] = p1[1];
    }
    acc0 = wmma_bf16(a, b0, acc0);
    acc1 = wmma_bf16(a, b1, acc1);
  }
  float c0a = c0[col], c0b = c0[col + 16];
#pragma unroll
  for (int r = 0; r < 8; ++r) {
    int gi = i0 + (half << 3) + r;
    float v0 = acc0[r] + c0a, v1 = acc1[r] + c0b;
    sf[((size_t)b * S + gi) * HD + col]      = v0;
    sf[((size_t)b * S + gi) * HD + col + 16] = v1;
    sft[((size_t)b * HD + col) * S + gi]      = (__bf16)v0;
    sft[((size_t)b * HD + col + 16) * S + gi] = (__bf16)v1;
  }
}

// ---------------- kernel 4: f1/f2 ---------------------------------------
__global__ void k_f12(const float* __restrict__ sf, const float* __restrict__ a1,
                      const float* __restrict__ a2, float* __restrict__ f1,
                      float* __restrict__ f2) {
  int idx = blockIdx.x * blockDim.x + threadIdx.x;   // < B*S
  const float* row = sf + (size_t)idx * HD;
  float s1 = 0.f, s2 = 0.f;
#pragma unroll
  for (int d = 0; d < HD; ++d) {
    float v = row[d];
    s1 += v * a1[d];
    s2 += v * a2[d];
  }
  f1[idx] = s1;
  f2[idx] = s2;
}

// ---------------- kernel 5: online column-softmax stats -----------------
// softmax over axis=1 (i): per (b,j) compute max_i and sum_i exp.
__global__ void __launch_bounds__(256)
k_stats(const float* __restrict__ bias, const float* __restrict__ f1,
        const float* __restrict__ f2, float* __restrict__ mcol,
        float* __restrict__ rscol) {
  int b = blockIdx.x >> 3;
  int j = ((blockIdx.x & 7) << 8) + threadIdx.x;
  const float* bb = bias + ((size_t)b * 4 + 3) * (size_t)S * S;
  float f2v = f2[b * S + j];
  float m = -1e30f, s = 0.f;
  for (int i = 0; i < S; ++i) {
    float x = leaky(f1[b * S + i] + f2v) + bb[(size_t)i * S + j];
    float nm = fmaxf(m, x);
    s = s * __expf(m - nm) + __expf(x - nm);
    m = nm;
  }
  mcol[b * S + j] = m;
  rscol[b * S + j] = 1.f / s;
}

// ---------------- kernel 6: out = tile(leaky(P @ seq_fts), 8) -----------
// bias tiles (16x32 f32) double-buffered in LDS via async-to-LDS; P tiles
// regenerated in registers (leaky/exp/scale) and fed straight into WMMA.
__global__ void __launch_bounds__(256)
k_out(const float* __restrict__ bias, const float* __restrict__ f1g,
      const float* __restrict__ f2g, const float* __restrict__ mg,
      const float* __restrict__ rsg, const __bf16* __restrict__ sft,
      float* __restrict__ out) {
  __shared__ float shf2[S], shm[S], shrs[S];         // 24 KB
  __shared__ float btile[8][2][16 * 32];             // 32 KB
  int lane = threadIdx.x & 31, wave = threadIdx.x >> 5;
  int tile = blockIdx.x * 8 + wave;
  int b = tile >> 7;                                  // all 8 waves same b
  int i0 = (tile & 127) << 4;
  for (int idx = threadIdx.x; idx < S; idx += 256) {
    shf2[idx] = f2g[b * S + idx];
    shm[idx]  = mg[b * S + idx];
    shrs[idx] = rsg[b * S + idx];
  }
  __syncthreads();
  int row = lane & 15, half = lane >> 4, col = lane & 15;
  float f1v = f1g[b * S + i0 + row];
  const float* biasb =
      bias + ((size_t)b * 4 + 3) * (size_t)S * S;
  int crow = lane >> 1, cpart = lane & 1;
  const float* gsrc = biasb + (size_t)(i0 + crow) * S + cpart * 16;
  float* ldst0 = &btile[wave][0][crow * 32 + cpart * 16];
  float* ldst1 = &btile[wave][1][crow * 32 + cpart * 16];
  const __bf16* sfb = sft + (size_t)b * HD * S;
  v8f acc0 = {}, acc1 = {};

  tile_copy64(gsrc, ldst0);
  for (int k0 = 0; k0 < S; k0 += 32) {
    if (k0 + 32 < S) {
      tile_copy64(gsrc + k0 + 32, (k0 & 32) ? ldst0 : ldst1);
      WAIT_ASYNC(4);
    } else {
      WAIT_ASYNC(0);
    }
    const float* cur = &btile[wave][(k0 >> 5) & 1][0];
    v16bf a, b0, b1;
#pragma unroll
    for (int v = 0; v < 8; ++v) {
      int kb = ((v >> 2) << 4) + (half << 3) + ((v & 3) << 1);
      int j = k0 + kb;
#pragma unroll
      for (int t = 0; t < 2; ++t) {
        float l = leaky(f1v + shf2[j + t]);
        float p = __expf(l + cur[row * 32 + kb + t] - shm[j + t]) * shrs[j + t];
        a[2 * v + t] = (__bf16)p;
      }
      int kk = k0 + ((v >> 2) << 4) + (half << 3) + ((v & 3) << 1);
      const __bf16* p0 = sfb + (size_t)col * S + kk;
      const __bf16* p1 = sfb + (size_t)(col + 16) * S + kk;
      b0[2 * v] = p0[0]; b0[2 * v + 1] = p0[1];
      b1[2 * v] = p1[0]; b1[2 * v + 1] = p1[1];
    }
    acc0 = wmma_bf16(a, b0, acc0);
    acc1 = wmma_bf16(a, b1, acc1);
  }
#pragma unroll
  for (int r = 0; r < 8; ++r) {
    int gi = i0 + (half << 3) + r;
    float v0 = leaky(acc0[r]), v1 = leaky(acc1[r]);
    float* orow = out + ((size_t)b * S + gi) * H;
#pragma unroll
    for (int hh = 0; hh < HEADS; ++hh) {
      orow[hh * HD + col]      = v0;
      orow[hh * HD + 16 + col] = v1;
    }
  }
}

// ---------------------------------------------------------------------------
extern "C" void kernel_launch(void* const* d_in, const int* in_sizes, int n_in,
                              void* d_out, int out_size, void* d_ws,
                              size_t ws_size, hipStream_t stream) {
  const float* inputs = (const float*)d_in[0];
  const float* graphs = (const float*)d_in[1];
  const float* biases = (const float*)d_in[2];
  const float* Wgc    = (const float*)d_in[3];
  const float* bgc    = (const float*)d_in[4];
  const float* W1     = (const float*)d_in[5];
  const float* a1     = (const float*)d_in[6];
  const float* a2     = (const float*)d_in[7];
  float* out = (float*)d_out;

  char* ws = (char*)d_ws;
  size_t off = 0;
  auto carve = [&](size_t bytes) -> void* {
    void* p = ws + off;
    off += (bytes + 255) & ~(size_t)255;
    return p;
  };
  float*  Wct  = (float*)carve((size_t)HD * H * 4);
  float*  c0   = (float*)carve(HD * 4);
  __bf16* Yt   = (__bf16*)carve((size_t)BB * HD * S * 2);
  float*  sf   = (float*)carve((size_t)BB * S * HD * 4);
  __bf16* sft  = (__bf16*)carve((size_t)BB * HD * S * 2);
  float*  f1   = (float*)carve((size_t)BB * S * 4);
  float*  f2   = (float*)carve((size_t)BB * S * 4);
  float*  mcol = (float*)carve((size_t)BB * S * 4);
  float*  rscl = (float*)carve((size_t)BB * S * 4);

  k_prep<<<dim3(1), dim3(256), 0, stream>>>(Wgc, bgc, W1, Wct, c0);
  k_y<<<dim3(64), dim3(256), 0, stream>>>(inputs, Wct, Yt);
  k_seqfts<<<dim3(64), dim3(256), 0, stream>>>(graphs, Yt, c0, sf, sft);
  k_f12<<<dim3(32), dim3(256), 0, stream>>>(sf, a1, a2, f1, f2);
  k_stats<<<dim3(32), dim3(256), 0, stream>>>(biases, f1, f2, mcol, rscl);
  k_out<<<dim3(64), dim3(256), 0, stream>>>(biases, f1, f2, mcol, rscl, sft, out);
}
